// RGN_62259845922891
// MI455X (gfx1250) — matrix operand order repaired and checked by reading, not compiled, not run on hardware
//
#include <hip/hip_runtime.h>
#include <hip/hip_bf16.h>

// ---------------- problem constants ----------------
#define L_SEQ   700
#define BATCH_N 32
#define D_IN    41
#define HID     800
#define G4      3200      // 4*HID
#define H2      1600      // 2*HID
#define LIN_N   20
#define LINPAD  128       // padded output cols for WMMA linear layer
#define ALPHA_N 20
#define ROWS    (L_SEQ*BATCH_N)   // 22400
#define K0PAD   64
#define NWG     25        // workgroups per direction in the scan (25*32 = 800 units)
#define HS_STRIDE 808     // padded LDS stride for h

typedef __bf16 bf16_t;
typedef __attribute__((ext_vector_type(16))) __bf16 v16bf;
typedef __attribute__((ext_vector_type(8)))  float  v8f;

__device__ __forceinline__ unsigned short f2bf(float f) {
  union { float f; unsigned u; } v; v.f = f;
  unsigned r = v.u + 0x7FFFu + ((v.u >> 16) & 1u);
  return (unsigned short)(r >> 16);
}
__device__ __forceinline__ float bf2f(unsigned short h) {
  union { unsigned u; float f; } v; v.u = ((unsigned)h) << 16;
  return v.f;
}

// ---------------- prep kernels ----------------
// x: [B, L, 41] f32  ->  A0: [L*B, 64] bf16 (zero padded K)
__global__ void pad_x_kernel(const float* __restrict__ x, unsigned short* __restrict__ dst) {
  int idx = blockIdx.x * blockDim.x + threadIdx.x;
  if (idx >= ROWS * K0PAD) return;
  int row = idx / K0PAD, k = idx % K0PAD;
  int t = row / BATCH_N, b = row % BATCH_N;
  float v = (k < D_IN) ? x[((size_t)b * L_SEQ + t) * D_IN + k] : 0.f;
  dst[idx] = f2bf(v);
}

// w_ih_l0: [2,3200,41] f32 -> [2,3200,64] bf16 (zero padded K)
__global__ void pad_w0_kernel(const float* __restrict__ w, unsigned short* __restrict__ dst) {
  int idx = blockIdx.x * blockDim.x + threadIdx.x;
  if (idx >= 2 * G4 * K0PAD) return;
  int k = idx % K0PAD;
  int row = idx / K0PAD;               // [2*3200)
  float v = (k < D_IN) ? w[(size_t)row * D_IN + k] : 0.f;
  dst[idx] = f2bf(v);
}

// w_lin: [20,1600] f32 -> [128,1600] bf16 (zero padded rows); also pad b_lin -> [128] f32
__global__ void pad_wlin_kernel(const float* __restrict__ w, const float* __restrict__ b,
                                unsigned short* __restrict__ dst, float* __restrict__ bpad) {
  int idx = blockIdx.x * blockDim.x + threadIdx.x;
  if (idx < LINPAD) bpad[idx] = (idx < LIN_N) ? b[idx] : 0.f;
  if (idx >= LINPAD * H2) return;
  int row = idx / H2, k = idx % H2;
  float v = (row < LIN_N) ? w[(size_t)row * H2 + k] : 0.f;
  dst[idx] = f2bf(v);
}

// generic f32 -> bf16
__global__ void cvt_bf16_kernel(const float* __restrict__ src, unsigned short* __restrict__ dst, int n) {
  for (int i = blockIdx.x * blockDim.x + threadIdx.x; i < n; i += gridDim.x * blockDim.x)
    dst[i] = f2bf(src[i]);
}

// ---------------- WMMA GEMM: C[M,N] = A[M,K]bf16 @ Bt[N,K]bf16^T + bias1 (+ bias2) ----------------
// grid = (N/128, M/32), block = 512 (16 waves, one 16x16 C tile per wave)
__global__ void __launch_bounds__(512) wmma_gemm_bias(
    const __bf16* __restrict__ A, int lda,
    const __bf16* __restrict__ Bt, int ldb,
    const float* __restrict__ bias1, const float* __restrict__ bias2,
    float* __restrict__ C, int ldc, int K)
{
  const int tid = threadIdx.x, lane = tid & 31, wv = tid >> 5;
  const int mt = wv >> 3, nt = wv & 7;
  const int m0 = blockIdx.y * 32, n0 = blockIdx.x * 128;
  const int mrow = m0 + mt * 16 + (lane & 15);
  const int ncol = n0 + nt * 16 + (lane & 15);
  const int abase = (lane < 16) ? 0 : 16;   // A: K rows 0..15 / 16..31
  const int bbase = (lane < 16) ? 0 : 16;   // B: K rows 0..15 / 16..31
  v8f acc = {};
  for (int kk = 0; kk < K; kk += 32) {
    v16bf af, bfv;
    const __bf16* ap = A + (size_t)mrow * lda + kk + abase;
#pragma unroll
    for (int i = 0; i < 8; ++i) { af[i] = ap[i]; af[8 + i] = ap[8 + i]; }
    const __bf16* bp = Bt + (size_t)ncol * ldb + kk + bbase;
#pragma unroll
    for (int i = 0; i < 16; ++i) bfv[i] = bp[i];
    acc = __builtin_amdgcn_wmma_f32_16x16x32_bf16(false, af, false, bfv, (short)0, acc, false, false);
  }
  const int rbase = m0 + mt * 16 + ((lane < 16) ? 0 : 8);
  const int col = n0 + nt * 16 + (lane & 15);
  const float bsum = bias1[col] + (bias2 ? bias2[col] : 0.f);
#pragma unroll
  for (int r = 0; r < 8; ++r)
    C[(size_t)(rbase + r) * ldc + col] = acc[r] + bsum;
}

// ---------------- persistent bidirectional LSTM scan ----------------
// grid = 2*NWG blocks of 512; dir = blockIdx.x / NWG (0 = fwd, 1 = bwd)
// Each WG owns 32 hidden units: W_hh slice (128x800 bf16) lives in LDS all 700 steps.
#define SCAN_SMEM (128*HID*2 + BATCH_N*HS_STRIDE*2 + BATCH_N*128*4 + BATCH_N*32*4)

__global__ void __launch_bounds__(512, 1) lstm_scan_kernel(
    const float* __restrict__ gates,        // [2][L*B][3200] f32 (x-projection + biases)
    const __bf16* __restrict__ whh,         // [2][3200][800] bf16
    unsigned short* __restrict__ hout,      // [L][B][1600] bf16 (fwd cols 0..799, bwd 800..1599)
    int* __restrict__ bar)                  // [2][L] arrive counters (pre-zeroed)
{
  extern __shared__ char smem[];
  __bf16* Bs = (__bf16*)smem;                                       // [128][800]
  __bf16* Hs = (__bf16*)(smem + 128 * HID * 2);                     // [32][808]
  float*  Gb = (float*)(smem + 128 * HID * 2 + BATCH_N * HS_STRIDE * 2); // [32][128]
  float*  Cb = Gb + BATCH_N * 128;                                  // [32][32]

  const int tid = threadIdx.x;
  const int wg = blockIdx.x;
  const int dir = wg / NWG;
  const int u0 = (wg % NWG) * 32;
  const int dirbase = dir * HID;
  const float* g_in = gates + (size_t)dir * ROWS * G4;
  const __bf16* w_in = whh + (size_t)dir * G4 * HID;
  int* my_bar = bar + dir * L_SEQ;

  // pin W_hh slice in LDS (rows ordered c = q*32+j  ->  W_hh row q*800+u0+j)
  const int NV4 = 128 * (HID / 8);
  for (int e = tid; e < NV4; e += 512) {
    int c = e / (HID / 8), k4 = e % (HID / 8);
    int row = ((c >> 5) * HID) + u0 + (c & 31);
    ((uint4*)(Bs + (size_t)c * HID))[k4] = ((const uint4*)(w_in + (size_t)row * HID))[k4];
  }
  for (int e = tid; e < BATCH_N * 32; e += 512) Cb[e] = 0.f;

  const int lane = tid & 31, wv = tid >> 5;
  const int mt = wv >> 3, nt = wv & 7;
  const int mrow = mt * 16 + (lane & 15);
  const int ncol = nt * 16 + (lane & 15);
  const int abase = (lane < 16) ? 0 : 16;
  const int bbase = (lane < 16) ? 0 : 16;

  for (int ts = 0; ts < L_SEQ; ++ts) {
    const int tt = dir ? (L_SEQ - 1 - ts) : ts;
    // stage h_{t-1} (full 800 units, all WGs) into LDS
    if (ts == 0) {
      uint4 z; z.x = z.y = z.z = z.w = 0u;
      for (int e4 = tid; e4 < BATCH_N * (HID / 8); e4 += 512) {
        int b = e4 / (HID / 8), k4 = e4 % (HID / 8);
        ((uint4*)(Hs + (size_t)b * HS_STRIDE))[k4] = z;
      }
    } else {
      const int tprev = dir ? (tt + 1) : (tt - 1);
      for (int e4 = tid; e4 < BATCH_N * (HID / 8); e4 += 512) {
        int b = e4 / (HID / 8), k4 = e4 % (HID / 8);
        ((uint4*)(Hs + (size_t)b * HS_STRIDE))[k4] =
            ((const uint4*)(hout + ((size_t)tprev * BATCH_N + b) * H2 + dirbase))[k4];
      }
    }
    __syncthreads();

    // h_{t-1} @ W_hh^T : each wave one 16x16 tile, K = 800 in 25 wmma
    v8f acc = {};
#pragma unroll 5
    for (int kk = 0; kk < HID; kk += 32) {
      v16bf af, bfv;
      const __bf16* ap = Hs + (size_t)mrow * HS_STRIDE + kk + abase;
#pragma unroll
      for (int i = 0; i < 8; ++i) { af[i] = ap[i]; af[8 + i] = ap[8 + i]; }
      const __bf16* bp = Bs + (size_t)ncol * HID + kk + bbase;
#pragma unroll
      for (int i = 0; i < 16; ++i) bfv[i] = bp[i];
      acc = __builtin_amdgcn_wmma_f32_16x16x32_bf16(false, af, false, bfv, (short)0, acc, false, false);
    }
    {
      const int rbase = mt * 16 + ((lane < 16) ? 0 : 8);
      const int col = nt * 16 + (lane & 15);
#pragma unroll
      for (int r = 0; r < 8; ++r) Gb[(rbase + r) * 128 + col] = acc[r];
    }
    __syncthreads();

    // gate nonlinearity + state update for this WG's 32 units
#pragma unroll
    for (int rep = 0; rep < 2; ++rep) {
      const int idx = tid + rep * 512;           // 1024 (b,u) pairs
      const int b = idx >> 5, u = idx & 31;
      const float* gx = g_in + ((size_t)tt * BATCH_N + b) * G4 + u0 + u;
      const float xi = gx[0]        + Gb[b * 128 +  0 + u];
      const float xf = gx[HID]      + Gb[b * 128 + 32 + u];
      const float xg = gx[2 * HID]  + Gb[b * 128 + 64 + u];
      const float xo = gx[3 * HID]  + Gb[b * 128 + 96 + u];
      const float ig = 1.f / (1.f + __expf(-xi));
      const float fg = 1.f / (1.f + __expf(-xf));
      const float gg = tanhf(xg);
      const float og = 1.f / (1.f + __expf(-xo));
      const float c = fg * Cb[b * 32 + u] + ig * gg;
      Cb[b * 32 + u] = c;
      hout[((size_t)tt * BATCH_N + b) * H2 + dirbase + u0 + u] = f2bf(og * tanhf(c));
    }
    __threadfence();
    __syncthreads();
    if (tid == 0) {
      __hip_atomic_fetch_add(my_bar + ts, 1, __ATOMIC_ACQ_REL, __HIP_MEMORY_SCOPE_AGENT);
      while (__hip_atomic_load(my_bar + ts, __ATOMIC_ACQUIRE, __HIP_MEMORY_SCOPE_AGENT) < NWG)
        __builtin_amdgcn_s_sleep(1);
    }
    __syncthreads();
  }
}

// ---------------- tail kernels ----------------
// softmax over the BATCH dim (faithful to reference): one wave per (t, o)
// logits are padded [row][128]; p is compact [row][20]
__global__ void softmax_batch_kernel(const float* __restrict__ logits, float* __restrict__ p) {
  int gid = blockIdx.x * blockDim.x + threadIdx.x;
  int widx = gid >> 5, lane = gid & 31;
  if (widx >= L_SEQ * LIN_N) return;
  int t = widx / LIN_N, o = widx % LIN_N;
  float v = logits[((size_t)t * BATCH_N + lane) * LINPAD + o];
  float m = v;
  for (int off = 16; off; off >>= 1) m = fmaxf(m, __shfl_xor(m, off, 32));
  float e = __expf(v - m);
  float s = e;
  for (int off = 16; off; off >>= 1) s += __shfl_xor(s, off, 32);
  p[((size_t)t * BATCH_N + lane) * LIN_N + o] = e / s;
}

// angles -> directly sin(phi), cos(phi): sin(atan2(s,c)) = s/hypot, cos = c/hypot
__global__ void angles_kernel(const float* __restrict__ p, const float* __restrict__ alphabet,
                              float* __restrict__ sphi, float* __restrict__ cphi) {
  int idx = blockIdx.x * blockDim.x + threadIdx.x;
  if (idx >= ROWS) return;
  const float* pp = p + (size_t)idx * ALPHA_N;
#pragma unroll
  for (int ax = 0; ax < 3; ++ax) {
    float s = 0.f, c = 0.f;
    for (int a = 0; a < ALPHA_N; ++a) {
      float w = pp[a], al = alphabet[a * 3 + ax];
      s += w * __sinf(al); c += w * __cosf(al);
    }
    float rh = 1.f / (sqrtf(s * s + c * c) + 1e-30f);
    sphi[(size_t)idx * 3 + ax] = s * rh;
    cphi[(size_t)idx * 3 + ax] = c * rh;
  }
}

// NeRF chain: one lane per batch element, 2100 strictly serial extension steps.
// flat(ang)[n*32+bb] ordering == flat [L][B][3] (the reference's reshape(L*3, B)).
__global__ void nerf_kernel(const float* __restrict__ sphi, const float* __restrict__ cphi,
                            float* __restrict__ out) {
  const int b = threadIdx.x;
  float ax = 0.f, ay = 0.f, az = 0.f;
  float bx = 100.f, by = 0.f, bz = 0.f;
  float cx = 200.f, cy = 100.f, cz = 0.f;
  const float BL[3] = {145.801f, 152.326f, 132.868f};
  const float BA[3] = {2.124f, 1.941f, 2.028f};
  float D0C[3], RST[3];
#pragma unroll
  for (int k = 0; k < 3; ++k) { D0C[k] = -BL[k] * cosf(BA[k]); RST[k] = BL[k] * sinf(BA[k]); }
  for (int n = 0; n < 3 * L_SEQ; ++n) {
    const float d0 = D0C[n % 3], rst = RST[n % 3];
    const float sp = sphi[(size_t)n * BATCH_N + b];
    const float cp = cphi[(size_t)n * BATCH_N + b];
    float ux = cx - bx, uy = cy - by, uz = cz - bz;
    float inv = 1.f / (sqrtf(ux * ux + uy * uy + uz * uz) + 1e-12f);
    const float bcx = ux * inv, bcy = uy * inv, bcz = uz * inv;
    const float vx = bx - ax, vy = by - ay, vz = bz - az;
    float nx = vy * bcz - vz * bcy;
    float ny = vz * bcx - vx * bcz;
    float nz = vx * bcy - vy * bcx;
    inv = 1.f / (sqrtf(nx * nx + ny * ny + nz * nz) + 1e-12f);
    nx *= inv; ny *= inv; nz *= inv;
    const float mx = ny * bcz - nz * bcy;
    const float my = nz * bcx - nx * bcz;
    const float mz = nx * bcy - ny * bcx;
    const float d1 = rst * cp;
    const float d2 = rst * sp;
    const float Dx = cx + d0 * bcx + d1 * mx + d2 * nx;
    const float Dy = cy + d0 * bcy + d1 * my + d2 * ny;
    const float Dz = cz + d0 * bcz + d1 * mz + d2 * nz;
    const size_t o = ((size_t)n * BATCH_N + b) * 3;
    out[o] = Dx; out[o + 1] = Dy; out[o + 2] = Dz;
    ax = bx; ay = by; az = bz; bx = cx; by = cy; bz = cz; cx = Dx; cy = Dy; cz = Dz;
  }
}

// ---------------- host ----------------
extern "C" void kernel_launch(void* const* d_in, const int* in_sizes, int n_in,
                              void* d_out, int out_size, void* d_ws, size_t ws_size,
                              hipStream_t stream) {
  const float* x    = (const float*)d_in[0];
  const float* wih0 = (const float*)d_in[1];
  const float* whh0 = (const float*)d_in[2];
  const float* bih0 = (const float*)d_in[3];
  const float* bhh0 = (const float*)d_in[4];
  const float* wih1 = (const float*)d_in[5];
  const float* whh1 = (const float*)d_in[6];
  const float* bih1 = (const float*)d_in[7];
  const float* bhh1 = (const float*)d_in[8];
  const float* wlin = (const float*)d_in[9];
  const float* blin = (const float*)d_in[10];
  const float* alph = (const float*)d_in[11];

  char* ws = (char*)d_ws;
  size_t off = 0;
  auto alloc = [&](size_t bytes) { char* p = ws + off; off += (bytes + 255) & ~(size_t)255; return p; };
  unsigned short* xbf    = (unsigned short*)alloc((size_t)ROWS * K0PAD * 2);
  unsigned short* w0bf   = (unsigned short*)alloc((size_t)2 * G4 * K0PAD * 2);
  unsigned short* w1bf   = (unsigned short*)alloc((size_t)2 * G4 * H2 * 2);
  unsigned short* wh0bf  = (unsigned short*)alloc((size_t)2 * G4 * HID * 2);
  unsigned short* wh1bf  = (unsigned short*)alloc((size_t)2 * G4 * HID * 2);
  unsigned short* wlbf   = (unsigned short*)alloc((size_t)LINPAD * H2 * 2);
  float*          blpad  = (float*)alloc((size_t)LINPAD * 4);
  float*          gates  = (float*)alloc((size_t)2 * ROWS * G4 * 4);   // reused by both layers
  unsigned short* h0     = (unsigned short*)alloc((size_t)ROWS * H2 * 2);
  unsigned short* h1     = (unsigned short*)alloc((size_t)ROWS * H2 * 2);
  float*          logitp = (float*)alloc((size_t)ROWS * LINPAD * 4);
  float*          pbuf   = (float*)alloc((size_t)ROWS * LIN_N * 4);
  float*          sphi   = (float*)alloc((size_t)ROWS * 3 * 4);
  float*          cphi   = (float*)alloc((size_t)ROWS * 3 * 4);
  int*            bar    = (int*)alloc((size_t)2 * L_SEQ * 4);

  hipFuncSetAttribute((const void*)lstm_scan_kernel,
                      hipFuncAttributeMaxDynamicSharedMemorySize, SCAN_SMEM);

  // -- prep / conversions --
  pad_x_kernel<<<(ROWS * K0PAD + 255) / 256, 256, 0, stream>>>(x, xbf);
  pad_w0_kernel<<<(2 * G4 * K0PAD + 255) / 256, 256, 0, stream>>>(wih0, w0bf);
  pad_wlin_kernel<<<(LINPAD * H2 + 255) / 256, 256, 0, stream>>>(wlin, blin, wlbf, blpad);
  cvt_bf16_kernel<<<4096, 256, 0, stream>>>(wih1, w1bf, 2 * G4 * H2);
  cvt_bf16_kernel<<<4096, 256, 0, stream>>>(whh0, wh0bf, 2 * G4 * HID);
  cvt_bf16_kernel<<<4096, 256, 0, stream>>>(whh1, wh1bf, 2 * G4 * HID);

  // -- layer 0: input projections (WMMA) + scan --
  for (int d = 0; d < 2; ++d)
    wmma_gemm_bias<<<dim3(G4 / 128, ROWS / 32), 512, 0, stream>>>(
        (const __bf16*)xbf, K0PAD,
        (const __bf16*)(w0bf + (size_t)d * G4 * K0PAD), K0PAD,
        bih0 + (size_t)d * G4, bhh0 + (size_t)d * G4,
        gates + (size_t)d * ROWS * G4, G4, K0PAD);
  hipMemsetAsync(bar, 0, 2 * L_SEQ * sizeof(int), stream);
  lstm_scan_kernel<<<2 * NWG, 512, SCAN_SMEM, stream>>>(gates, (const __bf16*)wh0bf, h0, bar);

  // -- layer 1: input projections (WMMA, K=1600) + scan --
  for (int d = 0; d < 2; ++d)
    wmma_gemm_bias<<<dim3(G4 / 128, ROWS / 32), 512, 0, stream>>>(
        (const __bf16*)h0, H2,
        (const __bf16*)(w1bf + (size_t)d * G4 * H2), H2,
        bih1 + (size_t)d * G4, bhh1 + (size_t)d * G4,
        gates + (size_t)d * ROWS * G4, G4, H2);
  hipMemsetAsync(bar, 0, 2 * L_SEQ * sizeof(int), stream);
  lstm_scan_kernel<<<2 * NWG, 512, SCAN_SMEM, stream>>>(gates, (const __bf16*)wh1bf, h1, bar);

  // -- linear layer via WMMA (N padded 20->128) --
  wmma_gemm_bias<<<dim3(LINPAD / 128, ROWS / 32), 512, 0, stream>>>(
      (const __bf16*)h1, H2, (const __bf16*)wlbf, H2,
      blpad, (const float*)nullptr, logitp, LINPAD, H2);

  // -- tail --
  softmax_batch_kernel<<<(L_SEQ * LIN_N * 32 + 255) / 256, 256, 0, stream>>>(logitp, pbuf);
  angles_kernel<<<(ROWS + 255) / 256, 256, 0, stream>>>(pbuf, alph, sphi, cphi);
  nerf_kernel<<<1, 32, 0, stream>>>(sphi, cphi, (float*)d_out);
}
